// SRUOneDirOnly_83614423318682
// MI455X (gfx1250) — matrix operand order repaired and checked by reading, not compiled, not run on hardware
//
#include <hip/hip_runtime.h>
#include <hip/hip_bf16.h>

typedef __attribute__((ext_vector_type(16))) __bf16 v16bf;
typedef __attribute__((ext_vector_type(8)))  __bf16 v8bf;
typedef __attribute__((ext_vector_type(4)))  __bf16 v4bf;
typedef __attribute__((ext_vector_type(8)))  float  v8f;
typedef __attribute__((ext_vector_type(4)))  float  v4f;

#define D_   1024
#define N3   3072
#define BB   32
#define LSEQ 512
#define LDSS 40   // padded LDS row stride in bf16 elements (bank-conflict-free)

// ---------------------------------------------------------------------------
// CDNA5 async global->LDS copy (ASYNCcnt-tracked), 16 bytes per lane.
// LDS destination offset = low 32 bits of the generic shared-space address
// (ISA 10.2: LDS aperture truncates to addr[31:0]).
// ---------------------------------------------------------------------------
__device__ __forceinline__ void async_copy_b128(void* lds_dst, const void* gsrc) {
  uint32_t loff = (uint32_t)(uintptr_t)lds_dst;
  asm volatile("global_load_async_to_lds_b128 %0, %1, off"
               :: "v"(loff), "v"(gsrc) : "memory");
}
__device__ __forceinline__ void wait_async0() {
  asm volatile("s_wait_asynccnt 0x0" ::: "memory");
}

// ---------------------------------------------------------------------------
// Split W (fp32, [4][D][3D]) into bf16 hi/lo, transposed to [4][N3][D]
// (K contiguous) so the GEMM streams K-major rows. 32x32 LDS tile transpose.
// ---------------------------------------------------------------------------
__global__ void convert_w_kernel(const float* __restrict__ W,
                                 __bf16* __restrict__ Wth,
                                 __bf16* __restrict__ Wtl) {
  __shared__ float tile[32][33];
  const int j  = blockIdx.z;
  const int n0 = blockIdx.x * 32;
  const int k0 = blockIdx.y * 32;
  const float* Wj = W + (size_t)j * D_ * N3;
#pragma unroll
  for (int i = 0; i < 4; ++i) {
    int k = threadIdx.y + i * 8;
    tile[k][threadIdx.x] = Wj[(size_t)(k0 + k) * N3 + (n0 + threadIdx.x)];
  }
  __syncthreads();
  __bf16* oh = Wth + (size_t)j * N3 * D_;
  __bf16* ol = Wtl + (size_t)j * N3 * D_;
#pragma unroll
  for (int i = 0; i < 4; ++i) {
    int n = threadIdx.y + i * 8;
    float w = tile[threadIdx.x][n];
    __bf16 hv = (__bf16)w;
    __bf16 lv = (__bf16)(w - (float)hv);
    oh[(size_t)(n0 + n) * D_ + (k0 + threadIdx.x)] = hv;
    ol[(size_t)(n0 + n) * D_ + (k0 + threadIdx.x)] = lv;
  }
}

// ---------------------------------------------------------------------------
// Split activations (fp32, M x K row-major) into bf16 hi/lo (same layout).
// ---------------------------------------------------------------------------
__global__ void convert_a_kernel(const float* __restrict__ in,
                                 __bf16* __restrict__ oh,
                                 __bf16* __restrict__ ol, int n4) {
  int i = blockIdx.x * blockDim.x + threadIdx.x;
  if (i >= n4) return;
  v4f x = ((const v4f*)in)[i];
  v4bf h, l;
#pragma unroll
  for (int c = 0; c < 4; ++c) {
    float xv = x[c];
    __bf16 hv = (__bf16)xv;
    h[c] = hv;
    l[c] = (__bf16)(xv - (float)hv);
  }
  ((v4bf*)oh)[i] = h;
  ((v4bf*)ol)[i] = l;
}

// ---------------------------------------------------------------------------
// U = A x B GEMM with split-bf16 (3 products) and f32 accumulate.
//   A:  M x 1024 (bf16 hi/lo, row-major)        M multiple of 128
//   B:  3072 x 1024 (bf16 hi/lo, N-major/K-contig)
//   U:  M x 3072 fp32
// Block: 128x128 tile, 256 threads = 8 waves (2x4), wave tile 64x32,
// BK = 32 -> v_wmma_f32_16x16x32_bf16, 24 WMMA / wave / K-step.
// Double-buffered LDS filled by async global->LDS DMA (ASYNCcnt).
// ---------------------------------------------------------------------------
__launch_bounds__(256, 1)
__global__ void sru_gemm_kernel(const __bf16* __restrict__ Ah,
                                const __bf16* __restrict__ Al,
                                const __bf16* __restrict__ Bh,
                                const __bf16* __restrict__ Bl,
                                float* __restrict__ U, int M) {
  // [buffer][matrix: Ah,Al,Bh,Bl][128 rows x LDSS]
  __shared__ __bf16 smem[2][4][128 * LDSS];

  const int tid  = threadIdx.x;
  const int bm   = blockIdx.x, bn = blockIdx.y;
  const int wave = tid >> 5;
  const int lane = tid & 31;
  const int l15  = lane & 15;
  const int hf   = lane >> 4;                  // which 16-lane half
  const int mbase = (wave >> 2) * 64;
  const int nbase = (wave & 3) * 32;

  const __bf16* gsrc[4] = {Ah, Al, Bh, Bl};

  v8f acc[4][2];
#pragma unroll
  for (int mt = 0; mt < 4; ++mt)
#pragma unroll
    for (int nt = 0; nt < 2; ++nt) acc[mt][nt] = (v8f)0.0f;

  // Stage one 4-slab set (A hi/lo + B hi/lo, 128 x 32 bf16 each) via async DMA.
  auto stage = [&](int dbuf, int kk) {
    for (int p = tid; p < 512; p += 256) {
      const int sel = p >> 7;                  // 0..3
      const int row = p & 127;
      const int grow = (sel < 2 ? bm : bn) * 128 + row;
      const char* src = (const char*)(gsrc[sel] + (size_t)grow * 1024 + kk);
      __bf16* dst = &smem[dbuf][sel][row * LDSS];
#pragma unroll
      for (int q = 0; q < 4; ++q)
        async_copy_b128((char*)dst + q * 16, src + q * 16);
    }
  };

  stage(0, 0);                                 // prologue: fill buffer 0

  for (int i = 0; i < 1024 / 32; ++i) {
    const int cur = i & 1;
    wait_async0();                             // this wave's fill of buf[cur] done
    __syncthreads();                           // all waves' fills visible; prev reads done
    if (i + 1 < 1024 / 32) stage(cur ^ 1, (i + 1) * 32);  // overlap DMA with compute

    // Fragment loads matching the 16-bit A-matrix VGPR layout:
    // lane half 0 holds K {0..7, 16..23}; half 1 holds K {8..15, 24..31}.
    v16bf fah[4], fal[4], fbh[2], fbl[2];
#pragma unroll
    for (int mt = 0; mt < 4; ++mt) {
      const __bf16* b0 = &smem[cur][0][(mbase + mt * 16 + l15) * LDSS];
      v8bf r1 = *(const v8bf*)(b0 + hf * 8);
      v8bf r2 = *(const v8bf*)(b0 + 16 + hf * 8);
      fah[mt] = __builtin_shufflevector(r1, r2, 0,1,2,3,4,5,6,7,8,9,10,11,12,13,14,15);
      const __bf16* b1 = &smem[cur][1][(mbase + mt * 16 + l15) * LDSS];
      r1 = *(const v8bf*)(b1 + hf * 8);
      r2 = *(const v8bf*)(b1 + 16 + hf * 8);
      fal[mt] = __builtin_shufflevector(r1, r2, 0,1,2,3,4,5,6,7,8,9,10,11,12,13,14,15);
    }
#pragma unroll
    for (int nt = 0; nt < 2; ++nt) {
      const __bf16* b0 = &smem[cur][2][(nbase + nt * 16 + l15) * LDSS + hf * 8];
      fbh[nt] = __builtin_shufflevector(*(const v8bf*)b0, *(const v8bf*)(b0 + 16),
                                        0,1,2,3,4,5,6,7,8,9,10,11,12,13,14,15);
      const __bf16* b1 = &smem[cur][3][(nbase + nt * 16 + l15) * LDSS + hf * 8];
      fbl[nt] = __builtin_shufflevector(*(const v8bf*)b1, *(const v8bf*)(b1 + 16),
                                        0,1,2,3,4,5,6,7,8,9,10,11,12,13,14,15);
    }

    // Split-bf16: hi*hi + hi*lo + lo*hi, fp32 accumulate.
#pragma unroll
    for (int mt = 0; mt < 4; ++mt)
#pragma unroll
      for (int nt = 0; nt < 2; ++nt) {
        acc[mt][nt] = __builtin_amdgcn_wmma_f32_16x16x32_bf16(
            false, fah[mt], false, fbh[nt], (short)0, acc[mt][nt], false, false);
        acc[mt][nt] = __builtin_amdgcn_wmma_f32_16x16x32_bf16(
            false, fah[mt], false, fbl[nt], (short)0, acc[mt][nt], false, false);
        acc[mt][nt] = __builtin_amdgcn_wmma_f32_16x16x32_bf16(
            false, fal[mt], false, fbh[nt], (short)0, acc[mt][nt], false, false);
      }
  }

  // C/D layout: VGPR i -> row i (lanes 0-15) / row 8+i (lanes 16-31), col = lane&15.
#pragma unroll
  for (int mt = 0; mt < 4; ++mt)
#pragma unroll
    for (int nt = 0; nt < 2; ++nt)
#pragma unroll
      for (int i = 0; i < 8; ++i) {
        int row = bm * 128 + mbase + mt * 16 + hf * 8 + i;
        int col = bn * 128 + nbase + nt * 16 + l15;
        U[(size_t)row * N3 + col] = acc[mt][nt][i];
      }
}

// ---------------------------------------------------------------------------
// Sequential SRU scan over Lc steps. One thread per (b,d); c in registers.
// Loads for step t+1 are independent of c -> issued before step-t math.
// ---------------------------------------------------------------------------
__global__ void sru_scan_kernel(const float* __restrict__ U,
                                const float* __restrict__ xin,
                                float* __restrict__ hout,
                                const float* __restrict__ vj,
                                const float* __restrict__ bj,
                                float* __restrict__ cbuf, int Lc) {
  const int g = blockIdx.x * blockDim.x + threadIdx.x;   // 0 .. B*D-1
  const int d = g & (D_ - 1);
  const int b = g >> 10;
  const float vf = vj[d], vr = vj[D_ + d];
  const float bf = bj[d], br = bj[D_ + d];
  const float SQ3 = 1.7320508075688772f;
  float c = cbuf[g];

  size_t ubase = (size_t)b * N3 + d;
  size_t xbase = (size_t)b * D_ + d;
  float u0 = U[ubase], uf = U[ubase + D_], ur = U[ubase + 2 * D_];
  float xt = xin[xbase];
  for (int t = 0; t < Lc; ++t) {
    float u0c = u0, ufc = uf, urc = ur, xc = xt;
    if (t + 1 < Lc) {                       // prefetch next step
      size_t un = ubase + (size_t)(t + 1) * BB * N3;
      u0 = U[un]; uf = U[un + D_]; ur = U[un + 2 * D_];
      xt = xin[xbase + (size_t)(t + 1) * BB * D_];
    }
    float f  = 1.0f / (1.0f + __expf(-(ufc + vf * c + bf)));
    float r  = 1.0f / (1.0f + __expf(-(urc + vr * c + br)));
    float cn = f * c + (1.0f - f) * u0c;
    hout[xbase + (size_t)t * BB * D_] = r * cn + (1.0f - f) * xc * SQ3;
    c = cn;
  }
  cbuf[g] = c;
}

// ---------------------------------------------------------------------------
extern "C" void kernel_launch(void* const* d_in, const int* in_sizes, int n_in,
                              void* d_out, int out_size, void* d_ws, size_t ws_size,
                              hipStream_t stream) {
  const float* x  = (const float*)d_in[0];   // (512,32,1024)
  const float* W  = (const float*)d_in[1];   // (4,1024,3072)
  const float* v  = (const float*)d_in[2];   // (4,2048)
  const float* bb = (const float*)d_in[3];   // (4,2048)
  const float* h0 = (const float*)d_in[4];   // (4,32,1024)
  float* out_h = (float*)d_out;                          // h: 512*32*1024
  float* out_c = out_h + (size_t)LSEQ * BB * D_;         // c: 4*32*1024

  // Workspace carve: split-W (fixed) + split-A + U (chunked by Lc).
  char* wp = (char*)d_ws;
  auto carve = [&](size_t bytes) -> char* {
    char* r = wp; wp += (bytes + 255) & ~(size_t)255; return r;
  };
  const size_t wsplit = (size_t)4 * N3 * D_ * sizeof(__bf16);
  __bf16* Wth = (__bf16*)carve(wsplit);
  __bf16* Wtl = (__bf16*)carve(wsplit);
  long avail = (long)ws_size - (long)(wp - (char*)d_ws);
  int Lc = LSEQ;                              // bytes/step: A hi+lo + U
  const long perL = (long)BB * D_ * 2 * 2 + (long)BB * N3 * 4;  // 524288
  while (Lc > 4 && (long)Lc * perL + 4096 > avail) Lc >>= 1;
  __bf16* Ah = (__bf16*)carve((size_t)Lc * BB * D_ * sizeof(__bf16));
  __bf16* Al = (__bf16*)carve((size_t)Lc * BB * D_ * sizeof(__bf16));
  float*  U  = (float*)carve((size_t)Lc * BB * N3 * sizeof(float));

  convert_w_kernel<<<dim3(N3 / 32, D_ / 32, 4), dim3(32, 8), 0, stream>>>(W, Wth, Wtl);

  for (int j = 0; j < 4; ++j) {
    const float* act_in = (j == 0) ? x : out_h;     // in-place h across layers
    hipMemcpyAsync(out_c + (size_t)j * BB * D_, h0 + (size_t)j * BB * D_,
                   (size_t)BB * D_ * sizeof(float), hipMemcpyDeviceToDevice, stream);
    for (int t0 = 0; t0 < LSEQ; t0 += Lc) {
      int Lcur = (LSEQ - t0 < Lc) ? (LSEQ - t0) : Lc;
      size_t off = (size_t)t0 * BB * D_;
      int n4 = Lcur * BB * D_ / 4;
      convert_a_kernel<<<(n4 + 255) / 256, 256, 0, stream>>>(act_in + off, Ah, Al, n4);
      int M = Lcur * BB;
      sru_gemm_kernel<<<dim3(M / 128, N3 / 128), 256, 0, stream>>>(
          Ah, Al, Wth + (size_t)j * N3 * D_, Wtl + (size_t)j * N3 * D_, U, M);
      sru_scan_kernel<<<dim3(BB * D_ / 256), 256, 0, stream>>>(
          U, act_in + off, out_h + off, v + (size_t)j * 2 * D_,
          bb + (size_t)j * 2 * D_, out_c + (size_t)j * BB * D_, Lcur);
    }
  }
}
